// Siglip2Attention_75677323756150
// MI455X (gfx1250) — matrix operand order repaired and checked by reading, not compiled, not run on hardware
//
#include <hip/hip_runtime.h>

// ---------------------------------------------------------------------------
// Problem constants (match reference)
// ---------------------------------------------------------------------------
#define T_LEN 2048
#define D_DIM 1152
#define H_NUM 16
#define DH 72          // head dim
#define DH_HALF 36     // rope half
#define DHP 96         // padded head dim for QK^T (3 x K=32 WMMA steps)
#define DHV 80         // padded head dim for P*V   (5 x N=16 tiles)
#define BATCH 8        // B in reference -> cu_seqlens has 9 entries

#if defined(__has_builtin)
#  if __has_builtin(__builtin_amdgcn_tensor_load_to_lds) && \
      __has_builtin(__builtin_amdgcn_s_wait_tensorcnt)
#    define HAVE_TDM 1
#  endif
#endif
#ifndef HAVE_TDM
#  define HAVE_TDM 0
#endif

typedef __attribute__((ext_vector_type(16))) __bf16 v16bf;
typedef __attribute__((ext_vector_type(8)))  __bf16 bf16x8;
typedef __attribute__((ext_vector_type(8)))  float  v8f;
typedef __attribute__((ext_vector_type(4)))  float  f32x4;
typedef __attribute__((ext_vector_type(4)))  unsigned int u32x4;
typedef __attribute__((ext_vector_type(8)))  int    i32x8;
typedef __attribute__((ext_vector_type(4)))  int    i32x4;

__device__ __forceinline__ v8f wmma_bf16(v16bf a, v16bf b, v8f c) {
    // D = A(16x32 bf16) * B(32x16 bf16) + C(16x16 f32)
    return __builtin_amdgcn_wmma_f32_16x16x32_bf16(
        /*neg_a=*/false, a, /*neg_b=*/false, b,
        /*c_mod=*/(short)0, c, /*reuse_a=*/false, /*reuse_b=*/false);
}

__device__ __forceinline__ v16bf concat8(bf16x8 lo, bf16x8 hi) {
    return __builtin_shufflevector(lo, hi, 0, 1, 2, 3, 4, 5, 6, 7,
                                   8, 9, 10, 11, 12, 13, 14, 15);
}

// ---------------------------------------------------------------------------
// Segment ids: seg[t] = searchsorted(cu, t, 'right') - 1
// ---------------------------------------------------------------------------
__global__ void seg_kernel(const int* __restrict__ cu, int* __restrict__ seg) {
    int t = blockIdx.x * blockDim.x + threadIdx.x;
    if (t >= T_LEN) return;
    int s = 0;
#pragma unroll
    for (int i = 1; i <= BATCH; ++i)
        if (cu[i] <= t) s = i;
    seg[t] = s;
}

// ---------------------------------------------------------------------------
// fp32 -> bf16 conversion (hidden states, done once)
// ---------------------------------------------------------------------------
__global__ __launch_bounds__(256)
void f32_to_bf16_kernel(const float* __restrict__ in, __bf16* __restrict__ out,
                        int n) {
    int i = blockIdx.x * blockDim.x + threadIdx.x;
    if (i < n) out[i] = (__bf16)in[i];
}

// ---------------------------------------------------------------------------
// WMMA GEMM: out[m, n] = sum_k A[m, k] * W[n, k] + bias[n]
//   A: bf16 row-major (M x K); W: f32 row-major (N x K).
//   Block = 256 threads = 8 waves in a 2(M) x 4(N) grid; each wave owns a
//   2x2 grid of 16x16 accumulator tiles => block tile 64 x 128.
//   A tile (64x32 bf16): streamed global->LDS by the Tensor Data Mover,
//     double buffered; TDM pad_enable (16 DW interval, 4 DW pad) produces the
//     40-element row stride => bank-conflict-free ds_load_b128 fragments.
//   B tile (128x32): f32 loads + cvt to bf16, staged to LDS, double buffered.
// ---------------------------------------------------------------------------
template <typename OT>
__global__ __launch_bounds__(256)
void wmma_gemm_bias(const __bf16* __restrict__ A, const float* __restrict__ W,
                    const float* __restrict__ bias, OT* __restrict__ out,
                    int M, int N, int K) {
    __shared__ __bf16 Abuf[2][64][40];
    __shared__ __bf16 Bbuf[2][128][40];

    const int tid  = threadIdx.x;
    const int lane = tid & 31;
    const int wave = tid >> 5;
    const int wm   = wave >> 2;          // 0..1  (M direction)
    const int wn   = wave & 3;           // 0..3  (N direction)
    const int half = lane >> 4;          // 16-lane half
    const int mrow = lane & 15;
    const int m0 = blockIdx.x * 64;
    const int n0 = blockIdx.y * 128;

    // cooperative-load index precompute
    const int ar  = tid >> 2;            // 0..63   A row
    const int ac  = (tid & 3) * 8;       // 0,8,16,24
    const int br  = tid >> 1;            // 0..127  B row
    const int bc  = (tid & 1) * 16;      // 0,16

    const int nk = K / 32;
    const bool tdm_wave = (wave == 0);

    // ---- A tile staging: TDM descriptor issue (wave 0 only) or fallback ----
    auto stage_a = [&](int ki, int b) {
#if HAVE_TDM
        if (tdm_wave) {
            const unsigned lds_off =
                (unsigned)(size_t)&Abuf[b][0][0];       // low 32b of LDS aperture ptr
            const unsigned long long ga =
                (unsigned long long)(size_t)(A + (size_t)m0 * K + (size_t)ki * 32);
            u32x4 g0;
            g0[0] = 1u;                                 // count=1, no gather
            g0[1] = lds_off;                            // D#.lds_addr
            g0[2] = (unsigned)ga;                       // D#.global_addr[31:0]
            g0[3] = ((unsigned)(ga >> 32) & 0x01FFFFFFu) | 0x80000000u; // addr[56:32] | type=2
            i32x8 g1;
            g1[0] = (1 << 16)       // data_size = 2 bytes
                  | (1 << 20)       // pad_enable
                  | (3 << 22)       // pad_interval: 16 DWORDs (one 64B tile row)
                  | (3 << 25);      // pad_amount:   4 DWORDs (16B -> 40-elem stride)
            g1[1] = (int)((unsigned)(K & 0xFFFF) << 16);            // tensor_dim0 lo16
            g1[2] = (int)(((unsigned)(M & 0xFFFF) << 16) | ((unsigned)K >> 16)); // dim1 lo16 | dim0 hi16
            g1[3] = (int)((32u << 16) | ((unsigned)M >> 16));       // tile_dim0=32 | dim1 hi16
            g1[4] = 64;                                             // tile_dim1 = 64 rows
            g1[5] = K;                                              // tensor_dim0_stride lo32
            g1[6] = 0;
            g1[7] = 0;
            i32x4 gz4 = {};
            i32x8 gz8 = {};
            __builtin_amdgcn_tensor_load_to_lds(g0, g1, gz4, gz4, gz8, 0);
        }
#else
        *(bf16x8*)&Abuf[b][ar][ac] =
            *(const bf16x8*)(A + (size_t)(m0 + ar) * K + (size_t)ki * 32 + ac);
#endif
    };

    // ---- B tile staging: f32 -> bf16 convert, all 256 threads ----
    auto stage_b = [&](int ki, int b) {
        const float* src = W + (size_t)(n0 + br) * K + (size_t)ki * 32 + bc;
        __builtin_prefetch(src + 32, 0, 3);   // next K slab of W
        f32x4 s0 = *(const f32x4*)(src + 0);
        f32x4 s1 = *(const f32x4*)(src + 4);
        f32x4 s2 = *(const f32x4*)(src + 8);
        f32x4 s3 = *(const f32x4*)(src + 12);
        bf16x8 t0, t1;
#pragma unroll
        for (int i = 0; i < 4; ++i) {
            t0[i]     = (__bf16)s0[i];
            t0[4 + i] = (__bf16)s1[i];
            t1[i]     = (__bf16)s2[i];
            t1[4 + i] = (__bf16)s3[i];
        }
        *(bf16x8*)&Bbuf[b][br][bc]     = t0;
        *(bf16x8*)&Bbuf[b][br][bc + 8] = t1;
    };

    v8f acc[2][2] = {{{}, {}}, {{}, {}}};

    // prologue: tile 0 in flight
    stage_a(0, 0);
    stage_b(0, 0);

    for (int i = 0; i < nk; ++i) {
        const int cur = i & 1;
        const int nxt = cur ^ 1;
        if (i + 1 < nk) {          // start next tile while current finishes
            stage_a(i + 1, nxt);
            stage_b(i + 1, nxt);
        }
#if HAVE_TDM
        if (tdm_wave) {            // in-order TENSORcnt: <=1 left => tile i done
            if (i + 1 < nk)
                __builtin_amdgcn_s_wait_tensorcnt(1);
            else
                __builtin_amdgcn_s_wait_tensorcnt(0);
        }
#endif
        __syncthreads();           // publish A (TDM) + B (ds stores) of tile i

        // ---- fragment reads from LDS ----
        v16bf afr[2], bfr[2];
#pragma unroll
        for (int mt = 0; mt < 2; ++mt) {
            const int r = wm * 32 + mt * 16 + mrow;
            bf16x8 lo = *(const bf16x8*)&Abuf[cur][r][half * 8];
            bf16x8 hi = *(const bf16x8*)&Abuf[cur][r][16 + half * 8];
            afr[mt] = concat8(lo, hi);
        }
#pragma unroll
        for (int nt = 0; nt < 2; ++nt) {
            const int r = wn * 32 + nt * 16 + mrow;
            bf16x8 lo = *(const bf16x8*)&Bbuf[cur][r][half * 8];
            bf16x8 hi = *(const bf16x8*)&Bbuf[cur][r][16 + half * 8];
            bfr[nt] = concat8(lo, hi);
        }
        // ---- 4 WMMAs per K-step (2x operand reuse) ----
#pragma unroll
        for (int mt = 0; mt < 2; ++mt)
#pragma unroll
            for (int nt = 0; nt < 2; ++nt)
                acc[mt][nt] = wmma_bf16(afr[mt], bfr[nt], acc[mt][nt]);

        __syncthreads();           // reads of 'cur' done before it is re-filled
    }

    // ---- epilogue: C/D layout (M = r + 8*half, N = mrow) ----
#pragma unroll
    for (int mt = 0; mt < 2; ++mt)
#pragma unroll
        for (int nt = 0; nt < 2; ++nt) {
            const int col = n0 + wn * 32 + nt * 16 + mrow;
            const float bcol = bias[col];
#pragma unroll
            for (int r = 0; r < 8; ++r) {
                const int row = m0 + wm * 32 + mt * 16 + r + 8 * half;
                out[(size_t)row * N + col] = (OT)(acc[mt][nt][r] + bcol);
            }
        }
}

// ---------------------------------------------------------------------------
// RoPE + repack:
//   qpad/kpad : [T][H][DHP] bf16, rope applied, tail (72..95) zeroed
//   vt        : [H][DHV][T] bf16 transposed (contiguous along keys), tail zero
// ---------------------------------------------------------------------------
__global__ __launch_bounds__(256)
void rope_pack_kernel(const __bf16* __restrict__ q, const __bf16* __restrict__ k,
                      const __bf16* __restrict__ v,
                      const float* __restrict__ cosT, const float* __restrict__ sinT,
                      __bf16* __restrict__ qp, __bf16* __restrict__ kp,
                      __bf16* __restrict__ vt) {
    const int t = blockIdx.x;
    for (int idx = threadIdx.x; idx < H_NUM * DHP; idx += blockDim.x) {
        const int h = idx / DHP, c = idx % DHP;
        const size_t po = ((size_t)t * H_NUM + h) * DHP + c;
        if (c >= DH) {
            qp[po] = (__bf16)0.f;
            kp[po] = (__bf16)0.f;
            continue;
        }
        const size_t so = (size_t)t * D_DIM + h * DH + c;
        float qv = (float)q[so];
        float kv = (float)k[so];
        float qo, ko;
        if (c < DH_HALF) {
            float cs = cosT[(size_t)t * DH + c];
            float sn = sinT[(size_t)t * DH + c];
            float q2 = (float)q[so + DH_HALF];
            float k2 = (float)k[so + DH_HALF];
            qo = qv * cs - q2 * sn;
            ko = kv * cs - k2 * sn;
        } else {
            float cs = cosT[(size_t)t * DH + c - DH_HALF];
            float sn = sinT[(size_t)t * DH + c - DH_HALF];
            float q1 = (float)q[so - DH_HALF];
            float k1 = (float)k[so - DH_HALF];
            qo = qv * cs + q1 * sn;
            ko = kv * cs + k1 * sn;
        }
        qp[po] = (__bf16)qo;
        kp[po] = (__bf16)ko;
    }
    for (int idx = threadIdx.x; idx < H_NUM * DHV; idx += blockDim.x) {
        const int h = idx / DHV, c = idx % DHV;
        __bf16 val = (c < DH) ? v[(size_t)t * D_DIM + h * DH + c] : (__bf16)0.f;
        vt[((size_t)h * DHV + c) * T_LEN + t] = val;
    }
}

// ---------------------------------------------------------------------------
// Flash attention with segment (block-diagonal) mask, one wave per
// (16-query tile, head). Online softmax over 32-key chunks.
// ---------------------------------------------------------------------------
__global__ __launch_bounds__(32)
void attn_kernel(const __bf16* __restrict__ qpad, const __bf16* __restrict__ kpad,
                 const __bf16* __restrict__ vt, const int* __restrict__ seg,
                 __bf16* __restrict__ attn_out, float scale) {
    __shared__ __bf16 pl[16 * 32];   // P re-layout buffer (C-layout -> A-fragment)

    const int lane = threadIdx.x & 31;
    const int half = lane >> 4;
    const int mrow = lane & 15;
    const int tq = blockIdx.x * 16;  // query tile base
    const int h  = blockIdx.y;
    const float NEG = -__builtin_inff();

    // Q A-fragments for the 3 K=32 steps over padded head dim (96)
    v16bf aq[3];
#pragma unroll
    for (int ks = 0; ks < 3; ++ks) {
        const __bf16* ap = qpad + ((size_t)(tq + mrow) * H_NUM + h) * DHP + ks * 32 + half * 8;
        aq[ks] = concat8(*(const bf16x8*)ap, *(const bf16x8*)(ap + 16));
    }

    // Segment ids: per accumulator row (r + 8*half), and tile min/max for skipping
    int segq[8];
#pragma unroll
    for (int r = 0; r < 8; ++r) segq[r] = seg[tq + r + 8 * half];
    int sq_lo = seg[tq + mrow], sq_hi = sq_lo;
#pragma unroll
    for (int m = 8; m >= 1; m >>= 1) {
        int lo = __shfl_xor(sq_lo, m, 32), hi = __shfl_xor(sq_hi, m, 32);
        sq_lo = lo < sq_lo ? lo : sq_lo;
        sq_hi = hi > sq_hi ? hi : sq_hi;
    }

    float mx[8], lsum[8];
    v8f o[5];
#pragma unroll
    for (int r = 0; r < 8; ++r) { mx[r] = -3e38f; lsum[r] = 0.f; }
#pragma unroll
    for (int n = 0; n < 5; ++n) o[n] = (v8f){};

    for (int kb = 0; kb < T_LEN; kb += 32) {
        const int sk0 = seg[kb + mrow];        // keys kb + 0..15  (this lane's col)
        const int sk1 = seg[kb + 16 + mrow];   // keys kb + 16..31
        // skip fully-masked key chunks (wave-uniform)
        if (!__any((sk0 >= sq_lo && sk0 <= sq_hi) || (sk1 >= sq_lo && sk1 <= sq_hi)))
            continue;

        // ---- load ALL K fragments for this chunk up front (clauses, overlap),
        //      and prefetch the V rows the P*V stage will need ----
        v16bf bk[3][2];
#pragma unroll
        for (int ks = 0; ks < 3; ++ks) {
            const __bf16* bp0 = kpad + ((size_t)(kb + mrow) * H_NUM + h) * DHP + ks * 32 + half * 8;
            const __bf16* bp1 = bp0 + (size_t)16 * H_NUM * DHP;
            bk[ks][0] = concat8(*(const bf16x8*)bp0, *(const bf16x8*)(bp0 + 16));
            bk[ks][1] = concat8(*(const bf16x8*)bp1, *(const bf16x8*)(bp1 + 16));
        }
#pragma unroll
        for (int n = 0; n < 5; ++n)
            __builtin_prefetch(vt + ((size_t)h * DHV + n * 16 + mrow) * T_LEN + kb + half * 16, 0, 3);

        // ---- S = Q * K^T: 6 back-to-back WMMAs ----
        v8f s0 = {}, s1 = {};
#pragma unroll
        for (int ks = 0; ks < 3; ++ks) {
            s0 = wmma_bf16(aq[ks], bk[ks][0], s0);
            s1 = wmma_bf16(aq[ks], bk[ks][1], s1);
        }

        // ---- mask + online softmax update ----
#pragma unroll
        for (int r = 0; r < 8; ++r) {
            float v0 = (segq[r] == sk0) ? s0[r] * scale : NEG;
            float v1 = (segq[r] == sk1) ? s1[r] * scale : NEG;
            float rmax = fmaxf(v0, v1);
#pragma unroll
            for (int m = 8; m >= 1; m >>= 1) rmax = fmaxf(rmax, __shfl_xor(rmax, m, 32));
            float Mn = fmaxf(mx[r], rmax);
            float alpha = __expf(mx[r] - Mn);
            float p0 = __expf(v0 - Mn);
            float p1 = __expf(v1 - Mn);
            float rs = p0 + p1;
#pragma unroll
            for (int m = 8; m >= 1; m >>= 1) rs += __shfl_xor(rs, m, 32);
            lsum[r] = lsum[r] * alpha + rs;
            mx[r] = Mn;
#pragma unroll
            for (int n = 0; n < 5; ++n) o[n][r] *= alpha;
            s0[r] = p0;
            s1[r] = p1;
        }

        // ---- re-layout P (C layout) -> A fragment via LDS ----
#pragma unroll
        for (int r = 0; r < 8; ++r) {
            pl[(r + 8 * half) * 32 + mrow]      = (__bf16)s0[r];
            pl[(r + 8 * half) * 32 + 16 + mrow] = (__bf16)s1[r];
        }
        __syncthreads();
        v16bf pa;
        {
            const __bf16* pp = &pl[mrow * 32 + half * 8];
            pa = concat8(*(const bf16x8*)pp, *(const bf16x8*)(pp + 16));
        }
        __syncthreads();

        // ---- O += P * V over 5 head-dim tiles ----
#pragma unroll
        for (int n = 0; n < 5; ++n) {
            const __bf16* vp = vt + ((size_t)h * DHV + n * 16 + mrow) * T_LEN + kb + half * 8;
            v16bf bv = concat8(*(const bf16x8*)vp, *(const bf16x8*)(vp + 16));
            o[n] = wmma_bf16(pa, bv, o[n]);
        }
    }

    // ---- finalize: divide by row sums, write bf16 (only cols < 72) ----
    float inv[8];
#pragma unroll
    for (int r = 0; r < 8; ++r) inv[r] = 1.0f / lsum[r];
#pragma unroll
    for (int n = 0; n < 5; ++n) {
        const int c = n * 16 + mrow;
        if (c >= DH) continue;
#pragma unroll
        for (int r = 0; r < 8; ++r) {
            const int row = tq + r + 8 * half;
            attn_out[(size_t)row * D_DIM + h * DH + c] = (__bf16)(o[n][r] * inv[r]);
        }
    }
}

// ---------------------------------------------------------------------------
// Host-side launcher
// ---------------------------------------------------------------------------
extern "C" void kernel_launch(void* const* d_in, const int* in_sizes, int n_in,
                              void* d_out, int out_size, void* d_ws, size_t ws_size,
                              hipStream_t stream) {
    const float* hs   = (const float*)d_in[0];
    const int*   cu   = (const int*)  d_in[1];
    const float* cosT = (const float*)d_in[2];
    const float* sinT = (const float*)d_in[3];
    const float* Wq   = (const float*)d_in[4];
    const float* bq   = (const float*)d_in[5];
    const float* Wk   = (const float*)d_in[6];
    const float* bk   = (const float*)d_in[7];
    const float* Wv   = (const float*)d_in[8];
    const float* bv   = (const float*)d_in[9];
    const float* Wo   = (const float*)d_in[10];
    const float* bo   = (const float*)d_in[11];
    float* out = (float*)d_out;

    // Workspace carve-up (~41 MB, all bf16 except seg)
    __bf16* hsb  = (__bf16*)d_ws;                       // [T][D] bf16 hidden
    __bf16* qb   = hsb  + (size_t)T_LEN * D_DIM;        // [T][D]
    __bf16* kb   = qb   + (size_t)T_LEN * D_DIM;        // [T][D]
    __bf16* vb   = kb   + (size_t)T_LEN * D_DIM;        // [T][D]
    __bf16* qpad = vb   + (size_t)T_LEN * D_DIM;        // [T][H][96]
    __bf16* kpad = qpad + (size_t)T_LEN * H_NUM * DHP;  // [T][H][96]
    __bf16* vt   = kpad + (size_t)T_LEN * H_NUM * DHP;  // [H][80][T]
    __bf16* attn = vt   + (size_t)H_NUM * DHV * T_LEN;  // [T][D]
    int*    seg  = (int*)(attn + (size_t)T_LEN * D_DIM);

    const dim3 gemm_grid(T_LEN / 64, D_DIM / 128);      // 32 x 9
    const dim3 gemm_block(256);
    const int  nelem = T_LEN * D_DIM;

    seg_kernel<<<(T_LEN + 255) / 256, 256, 0, stream>>>(cu, seg);
    f32_to_bf16_kernel<<<(nelem + 255) / 256, 256, 0, stream>>>(hs, hsb, nelem);

    // Q/K/V projections (TDM-staged A, f32 W staged->bf16, f32 WMMA accumulate)
    wmma_gemm_bias<__bf16><<<gemm_grid, gemm_block, 0, stream>>>(
        hsb, Wq, bq, qb, T_LEN, D_DIM, D_DIM);
    wmma_gemm_bias<__bf16><<<gemm_grid, gemm_block, 0, stream>>>(
        hsb, Wk, bk, kb, T_LEN, D_DIM, D_DIM);
    wmma_gemm_bias<__bf16><<<gemm_grid, gemm_block, 0, stream>>>(
        hsb, Wv, bv, vb, T_LEN, D_DIM, D_DIM);

    // RoPE + pad/transpose repack
    rope_pack_kernel<<<T_LEN, 256, 0, stream>>>(qb, kb, vb, cosT, sinT, qpad, kpad, vt);

    // Flash attention with segment mask
    const float scale = 1.0f / sqrtf((float)DH);
    attn_kernel<<<dim3(T_LEN / 16, H_NUM), 32, 0, stream>>>(
        qpad, kpad, vt, seg, attn, scale);

    // Output projection (bf16 in -> f32 out)
    wmma_gemm_bias<float><<<gemm_grid, gemm_block, 0, stream>>>(
        attn, Wo, bo, out, T_LEN, D_DIM, D_DIM);

    (void)in_sizes; (void)n_in; (void)out_size; (void)ws_size;
}
